// RNN_89790586290625
// MI455X (gfx1250) — matrix operand order repaired and checked by reading, not compiled, not run on hardware
//
#include <hip/hip_runtime.h>

#define T_STEPS 2048
#define BATCH   64
#define HID     512
#define INP     512

typedef __bf16 bf16;
typedef __attribute__((ext_vector_type(16))) __bf16 v16bf;
typedef __attribute__((ext_vector_type(8)))  __bf16 v8bf;
typedef __attribute__((ext_vector_type(8)))  float  v8f;
typedef __attribute__((ext_vector_type(4)))  unsigned int v4u;
typedef __attribute__((ext_vector_type(8)))  int v8i;
typedef __attribute__((ext_vector_type(4)))  int v4i;

union AFrag { v16bf v; v8bf h[2]; };

// ---------------------------------------------------------------------------
// Kernel A: convert U and W (fp32 [512,512]) to bf16 once per call.
// ---------------------------------------------------------------------------
__global__ void cvt_uw(const float* __restrict__ U, const float* __restrict__ W,
                       bf16* __restrict__ Ubf, bf16* __restrict__ Wbf) {
    int i = blockIdx.x * blockDim.x + threadIdx.x;
    if (i < HID * INP) {
        Ubf[i] = (bf16)U[i];
        Wbf[i] = (bf16)W[i];
    }
}

// ---------------------------------------------------------------------------
// Kernel B: Zbf[T*B, H] = bf16( X[T*B, I] @ U^T )  -- time-parallel part.
// Memory-bound phase: X read 256MB + Zbf write 128MB ~= 17us at 23.3 TB/s.
// Per wave: 16-row M-panel x 8 N-tiles, f32 accum.  All 8 B-tiles of a
// K-step are loaded into distinct registers first (single clause + one wait),
// then 8 WMMAs issue back-to-back.
// ---------------------------------------------------------------------------
__global__ void __launch_bounds__(256) gemm_xu(const float* __restrict__ X,
                                               const bf16* __restrict__ Ubf,
                                               bf16* __restrict__ Zbf) {
    const int lane = threadIdx.x & 31;
    const int wid  = threadIdx.x >> 5;
    const int gw   = blockIdx.x * 8 + wid;   // global wave id
    const int mt   = gw >> 2;                // M-tile (16 rows of T*B), 0..8191
    const int nt0  = (gw & 3) * 8;           // base N-tile of 8-tile group
    const int l16  = lane & 15;
    const int lh   = lane >> 4;

    const float* Arow = X + (size_t)(mt * 16 + l16) * INP;

    const v8f zero = {0.f, 0.f, 0.f, 0.f, 0.f, 0.f, 0.f, 0.f};
    v8f c[8];
#pragma unroll
    for (int j = 0; j < 8; ++j) c[j] = zero;

#pragma unroll
    for (int ks = 0; ks < 16; ++ks) {
        const int kb = ks * 32;
        // A fragment: load fp32, convert to bf16 in-register
        v8f flo = *(const v8f*)(Arow + kb + lh * 16);
        v8f fhi = *(const v8f*)(Arow + kb + lh * 16 + 8);
        AFrag a;
#pragma unroll
        for (int i = 0; i < 8; ++i) {
            a.h[0][i] = (bf16)flo[i];
            a.h[1][i] = (bf16)fhi[i];
        }
        // Batch all 8 B tiles, then back-to-back WMMAs
        v16bf b[8];
#pragma unroll
        for (int j = 0; j < 8; ++j)
            b[j] = *(const v16bf*)(Ubf + (size_t)((nt0 + j) * 16 + l16) * INP + kb + lh * 16);
#pragma unroll
        for (int j = 0; j < 8; ++j)
            c[j] = __builtin_amdgcn_wmma_f32_16x16x32_bf16(
                false, a.v, false, b[j], (short)0, c[j], false, false);
    }

    // Store bf16 Z: D layout -> VGPR v, lane: row = v + 8*(lane/16), col = lane%16
#pragma unroll
    for (int j = 0; j < 8; ++j) {
        const int n = (nt0 + j) * 16 + l16;
#pragma unroll
        for (int v = 0; v < 8; ++v) {
            const size_t row = (size_t)mt * 16 + v + 8 * lh;
            Zbf[row * HID + n] = (bf16)c[j][v];
        }
    }
}

// ---------------------------------------------------------------------------
// TDM: async-load one Z step tile (64 rows x 512 bf16, row stride 512) from
// global into LDS at byte offset lds_byte_off.  D# per CDNA5 ISA:
//   group0: count=1 | lds_addr | global_addr[56:0] | type=2
//   group1: data_size=1(2B), tensor_dim0=512, tensor_dim1=64,
//           tile_dim0=512, tile_dim1=64, tensor_dim0_stride=512
// 6-arg builtin form for this toolchain (clang-23/therock-10.0):
//   (v4u g0, v8i g1, v4i g2, v4i g3, v8i g4, i32 cpol)
// Must be issued under a wave-uniform branch (TDM ignores EXEC).
// ---------------------------------------------------------------------------
__device__ __forceinline__ void tdm_load_z_tile(const bf16* gsrc, unsigned lds_byte_off) {
    unsigned long long ga = (unsigned long long)(uintptr_t)gsrc;
    v4u g0 = { 1u,                                   // count=1
               lds_byte_off,                         // lds_addr
               (unsigned)ga,                         // global_addr[31:0]
               (unsigned)((ga >> 32) & 0x01FFFFFFull) | 0x80000000u }; // [56:32] | type=2
    v8i g1 = { 0x00010000,   // data_size=1 (2 bytes)
               0x02000000,   // tensor_dim0 = 512 (low 16 in d1[31:16])
               0x00400000,   // tensor_dim1 = 64  (low 16 in d2[31:16])
               0x02000000,   // tile_dim0 = 512   (d3[31:16])
               64,           // tile_dim1 = 64    (d4[15:0])
               512,          // tensor_dim0_stride = 512 (d5[31:0])
               0, 0 };
    v4i g2 = { 0, 0, 0, 0 };
    v4i g3 = { 0, 0, 0, 0 };
    v8i g4 = { 0, 0, 0, 0, 0, 0, 0, 0 };
    __builtin_amdgcn_tensor_load_to_lds(g0, g1, g2, g3, g4, 0);
}

// ---------------------------------------------------------------------------
// Kernel C: persistent recurrence, one workgroup = 32 wave32 waves on one WGP.
//   LDS: h ping-pong (2x64KB) + Z ping-pong (2x64KB) = 256KB of 320KB.
//   Wave 0 TDM-prefetches Z[t+1] into LDS one step ahead; s_wait_tensorcnt(0)
//   + the per-step workgroup barrier publish it.  W fragments are explicitly
//   double-buffered across K-steps so WMMAs never stall on their own loads.
// ---------------------------------------------------------------------------
__global__ void __launch_bounds__(1024) rnn_steps(const float* __restrict__ h0,
                                                  const bf16* __restrict__ Zbf,
                                                  const bf16* __restrict__ Wbf,
                                                  float* __restrict__ out) {
    __shared__ bf16 smem[4 * BATCH * HID];   // [h0 | h1 | z0 | z1], 256 KB
    bf16* const hb0 = smem;
    bf16* const hb1 = smem + BATCH * HID;
    bf16* const zb0 = smem + 2 * BATCH * HID;
    bf16* const zb1 = smem + 3 * BATCH * HID;

    const int tid  = threadIdx.x;
    const int lane = tid & 31;
    const int w    = tid >> 5;       // 0..31

    // Initial: convert h0 into buffer 0; wave 0 kicks off TDM for Z[0].
#pragma unroll
    for (int i = 0; i < (BATCH * HID) / 1024; ++i) {
        const int idx = tid + i * 1024;
        hb0[idx] = (bf16)h0[idx];
    }
    if (w == 0) {
        tdm_load_z_tile(Zbf, (unsigned)(2 * BATCH * HID * sizeof(bf16)));
    }

    const int mt   = w & 3;          // M-tile: 16 of the 64 batch rows
    const int nt0  = (w >> 2) * 4;   // 4 consecutive N-tiles (64 of 512 cols)
    const int l16  = lane & 15;
    const int lh   = lane >> 4;
    const int mrow = mt * 16 + l16;

    for (int t = 0; t < T_STEPS; ++t) {
        const bf16* hc  = (t & 1) ? hb1 : hb0;
        bf16*       hn  = (t & 1) ? hb0 : hb1;
        const bf16* zsm = (t & 1) ? zb1 : zb0;

        if (w == 0) {
            __builtin_amdgcn_s_wait_tensorcnt(0);   // Z[t] resident in LDS
            if (t + 1 < T_STEPS)
                tdm_load_z_tile(Zbf + (size_t)(t + 1) * BATCH * HID,
                                (unsigned)((2 + ((t + 1) & 1)) * BATCH * HID * sizeof(bf16)));
            if (t + 2 < T_STEPS)                    // warm HBM->L2 for next TDM
                __builtin_prefetch(Zbf + (size_t)(t + 2) * BATCH * HID + lane * 1024, 0, 0);
        }
        __syncthreads();                            // publish Z[t] (and h from t-1)

        // Accumulators seeded with Z[t] tile from LDS (bf16 -> f32).
        v8f c[4];
#pragma unroll
        for (int j = 0; j < 4; ++j) {
            const int n = (nt0 + j) * 16 + l16;
#pragma unroll
            for (int v = 0; v < 8; ++v) {
                const int b = mt * 16 + v + 8 * lh;
                c[j][v] = (float)zsm[b * HID + n];
            }
        }

        // K loop with explicit double-buffering of the W (B-operand) tiles.
        v16bf b[4];
#pragma unroll
        for (int j = 0; j < 4; ++j)
            b[j] = *(const v16bf*)(Wbf + (size_t)((nt0 + j) * 16 + l16) * HID + lh * 16);

#pragma unroll
        for (int ks = 0; ks < 16; ++ks) {
            const int kb = ks * 32;
            AFrag a;                                 // h_{t-1} fragment from LDS
            a.h[0] = *(const v8bf*)(hc + mrow * HID + kb + lh * 16);
            a.h[1] = *(const v8bf*)(hc + mrow * HID + kb + lh * 16 + 8);

            v16bf bn[4];
            if (ks < 15) {
#pragma unroll
                for (int j = 0; j < 4; ++j)
                    bn[j] = *(const v16bf*)(Wbf + (size_t)((nt0 + j) * 16 + l16) * HID
                                            + kb + 32 + lh * 16);
            }
#pragma unroll
            for (int j = 0; j < 4; ++j)
                c[j] = __builtin_amdgcn_wmma_f32_16x16x32_bf16(
                    false, a.v, false, b[j], (short)0, c[j], false, false);
            if (ks < 15) {
#pragma unroll
                for (int j = 0; j < 4; ++j) b[j] = bn[j];
            }
        }

        // tanh, write fp32 output [B, T, H] and bf16 next-h into LDS
#pragma unroll
        for (int j = 0; j < 4; ++j) {
            const int n = (nt0 + j) * 16 + l16;
#pragma unroll
            for (int v = 0; v < 8; ++v) {
                const int b = mt * 16 + v + 8 * lh;
                const float val = tanhf(c[j][v]);
                out[((size_t)b * T_STEPS + t) * HID + n] = val;
                hn[b * HID + n] = (bf16)val;
            }
        }
        __syncthreads();   // h and z buffers may be reused next step
    }
}

// ---------------------------------------------------------------------------
extern "C" void kernel_launch(void* const* d_in, const int* in_sizes, int n_in,
                              void* d_out, int out_size, void* d_ws, size_t ws_size,
                              hipStream_t stream) {
    const float* h0 = (const float*)d_in[0];   // [B, H]
    const float* X  = (const float*)d_in[1];   // [T, B, I]
    const float* W  = (const float*)d_in[2];   // [H, H]
    const float* U  = (const float*)d_in[3];   // [H, I]
    float* out = (float*)d_out;                // [B, T, H]

    // Workspace: Ubf (512KB) | Wbf (512KB) | Zbf (128MB bf16), rewritten each call
    bf16* Ubf = (bf16*)d_ws;
    bf16* Wbf = Ubf + HID * INP;
    bf16* Zbf = (bf16*)((char*)d_ws + (size_t)2 * HID * INP * sizeof(bf16));

    cvt_uw<<<dim3((HID * INP + 255) / 256), dim3(256), 0, stream>>>(U, W, Ubf, Wbf);

    // (T*B/16) M-tiles x 4 N-groups = 32768 wave-tasks / 8 waves per block
    gemm_xu<<<dim3(4096), dim3(256), 0, stream>>>(X, Ubf, Zbf);

    // Persistent sequential recurrence: one WGP, 32 waves
    rnn_steps<<<dim3(1), dim3(1024), 0, stream>>>(h0, Zbf, Wbf, out);
}
